// MultiHeadAttention_63170378989867
// MI455X (gfx1250) — compile-verified, matrix operand order checked by reading, not run on hardware
//
#include <hip/hip_runtime.h>
#include <hip/hip_bf16.h>

// ---------------------------------------------------------------------------
// MHA forward for B=4, S=2048, D=1024, H=16, DEPTH=64 on gfx1250 (CDNA5).
// All matmuls via v_wmma_f32_16x16x32_bf16 (wave32, 16x16 tiles, K=32).
// Attention stages K chunks via GLOBAL_LOAD_ASYNC_TO_LDS_B128 (ASYNCcnt) and
// V chunks via TENSOR_LOAD_TO_LDS (TDM, TENSORcnt).
// ---------------------------------------------------------------------------

#define BATCH 4
#define SEQ   2048
#define DM    1024
#define NH    16
#define DEPTH 64

typedef __attribute__((ext_vector_type(16))) __bf16 v16bf;
typedef __attribute__((ext_vector_type(8)))  __bf16 v8bf;
typedef __attribute__((ext_vector_type(8)))  float  v8f;
typedef __attribute__((ext_vector_type(4)))  unsigned int v4u;
typedef __attribute__((ext_vector_type(8)))  int v8i;
typedef __attribute__((ext_vector_type(4)))  int v4i;

__device__ __forceinline__ v16bf cat16(v8bf lo, v8bf hi) {
  return __builtin_shufflevector(lo, hi, 0,1,2,3,4,5,6,7,8,9,10,11,12,13,14,15);
}

// Load 8 consecutive f32 and convert to 8 bf16 (two b128 loads).
__device__ __forceinline__ v8bf cvt8f(const float* __restrict__ p) {
  const float4* p4 = reinterpret_cast<const float4*>(p);
  float4 a = p4[0], b = p4[1];
  v8bf r;
  r[0]=(__bf16)a.x; r[1]=(__bf16)a.y; r[2]=(__bf16)a.z; r[3]=(__bf16)a.w;
  r[4]=(__bf16)b.x; r[5]=(__bf16)b.y; r[6]=(__bf16)b.z; r[7]=(__bf16)b.w;
  return r;
}

__device__ __forceinline__ v8bf ld8(const __bf16* __restrict__ p) {
  return *reinterpret_cast<const v8bf*>(p);
}

// Async 16B global -> LDS copy (per lane), tracked by ASYNCcnt.
__device__ __forceinline__ void async_ld16(unsigned lds_off, const void* gaddr) {
  unsigned long long ga = (unsigned long long)(size_t)gaddr;
  asm volatile("global_load_async_to_lds_b128 %0, %1, off"
               :: "v"(lds_off), "v"(ga) : "memory");
}
__device__ __forceinline__ void wait_async() {
  asm volatile("s_wait_asynccnt 0" ::: "memory");
}

// ---------------------------------------------------------------------------
// Kernel 0: pack the four weight matrices to bf16 once.
// ---------------------------------------------------------------------------
__global__ __launch_bounds__(256) void pack_kernel(
    const float* __restrict__ wq, const float* __restrict__ wk,
    const float* __restrict__ wv, const float* __restrict__ fc,
    __bf16* __restrict__ wqb, __bf16* __restrict__ wkb,
    __bf16* __restrict__ wvb, __bf16* __restrict__ fcb)
{
  const int z = blockIdx.y;
  const float* src = (z == 0) ? wq : (z == 1) ? wk : (z == 2) ? wv : fc;
  __bf16*      dst = (z == 0) ? wqb : (z == 1) ? wkb : (z == 2) ? wvb : fcb;
  size_t i = ((size_t)blockIdx.x * 256 + threadIdx.x) * 8;   // 8 elems/thread
  *reinterpret_cast<v8bf*>(dst + i) = cvt8f(src + i);
}

// ---------------------------------------------------------------------------
// Kernel 1: projections  Y = X @ W^T, split heads.  2-stage software pipeline:
// fragments for step n+1 load while step n's 4 WMMAs execute.
// ---------------------------------------------------------------------------
__global__ __launch_bounds__(256) void proj_kernel(
    const float* __restrict__ q, const float* __restrict__ k,
    const float* __restrict__ v, const __bf16* __restrict__ wq,
    const __bf16* __restrict__ wk, const __bf16* __restrict__ wv,
    __bf16* __restrict__ qh, __bf16* __restrict__ kh, __bf16* __restrict__ vT)
{
  const int lane = threadIdx.x & 31;
  const int wave = threadIdx.x >> 5;
  const int mt   = blockIdx.x * 8 + wave;   // 0..511  (B*S/16 row tiles)
  const int nt   = blockIdx.y;              // 0..15   (head)
  const int z    = blockIdx.z;              // 0=q 1=k 2=v

  const float*  X = (z == 0) ? q : (z == 1) ? k : v;
  const __bf16* W = (z == 0) ? wq : (z == 1) ? wk : wv;

  const int m0 = mt * 16;
  const int n0 = nt * DEPTH;
  const int c0 = (lane >> 4) * 8;

  const float*  Ap = X + (size_t)(m0 + (lane & 15)) * DM + c0;
  const __bf16* Bp[4];
#pragma unroll
  for (int j = 0; j < 4; ++j)
    Bp[j] = W + (size_t)(n0 + j * 16 + (lane & 15)) * DM + c0;

  v8f acc[4] = {};
  v16bf a0 = cat16(cvt8f(Ap), cvt8f(Ap + 16));
  v16bf b0[4];
#pragma unroll
  for (int j = 0; j < 4; ++j) b0[j] = cat16(ld8(Bp[j]), ld8(Bp[j] + 16));

  for (int k0 = 0; k0 < DM; k0 += 64) {
    const int k1 = k0 + 32;
    v16bf a1 = cat16(cvt8f(Ap + k1), cvt8f(Ap + k1 + 16));
    v16bf b1[4];
#pragma unroll
    for (int j = 0; j < 4; ++j)
      b1[j] = cat16(ld8(Bp[j] + k1), ld8(Bp[j] + k1 + 16));
#pragma unroll
    for (int j = 0; j < 4; ++j)
      acc[j] = __builtin_amdgcn_wmma_f32_16x16x32_bf16(
          false, a0, false, b0[j], (short)0, acc[j], false, false);

    const int k2 = (k0 + 64 < DM) ? k0 + 64 : 0;   // last prefetch is dummy
    a0 = cat16(cvt8f(Ap + k2), cvt8f(Ap + k2 + 16));
#pragma unroll
    for (int j = 0; j < 4; ++j)
      b0[j] = cat16(ld8(Bp[j] + k2), ld8(Bp[j] + k2 + 16));
#pragma unroll
    for (int j = 0; j < 4; ++j)
      acc[j] = __builtin_amdgcn_wmma_f32_16x16x32_bf16(
          false, a1, false, b1[j], (short)0, acc[j], false, false);
  }

  // Epilogue: scatter into head-split layouts (bf16).
  const int b    = m0 >> 11;            // /SEQ
  const int sbas = m0 & (SEQ - 1);
  const int h    = nt;
#pragma unroll
  for (int j = 0; j < 4; ++j) {
#pragma unroll
    for (int r = 0; r < 8; ++r) {
      int mr  = r + ((lane >> 4) << 3);
      int s   = sbas + mr;
      int dep = j * 16 + (lane & 15);
      __bf16 val = (__bf16)acc[j][r];
      if (z == 0)
        qh[(((size_t)b * NH + h) * SEQ + s) * DEPTH + dep] = val;
      else if (z == 1)
        kh[(((size_t)b * NH + h) * SEQ + s) * DEPTH + dep] = val;
      else
        vT[(((size_t)b * NH + h) * DEPTH + dep) * SEQ + s] = val;
    }
  }
}

// ---------------------------------------------------------------------------
// Kernel 2: flash attention. 8 waves/block share one (b,h); each wave owns a
// 16-query tile. Per 32-key chunk: K staged via per-lane async b128 copies
// (ASYNCcnt); V tile (64 depth-rows x 32 keys, stride S) staged via one TDM
// TENSOR_LOAD_TO_LDS issued by wave 0 (TENSORcnt). P round-trips LDS to turn
// the f32 C/D layout into the bf16 A-operand layout.
// ---------------------------------------------------------------------------
__global__ __launch_bounds__(256) void attn_kernel(
    const __bf16* __restrict__ qh, const __bf16* __restrict__ kh,
    const __bf16* __restrict__ vT, const int* __restrict__ mask,
    __bf16* __restrict__ ctx)
{
  __shared__ __align__(16) __bf16 ksm[32 * 64];        // [key][depth]  4KB
  __shared__ __align__(16) __bf16 vsm[64 * 32];        // [depth][key]  4KB
  __shared__ __align__(16) __bf16 pshare[8][16 * 32];  // per-wave P    8KB

  const int t    = threadIdx.x;
  const int lane = t & 31;
  const int wave = t >> 5;
  const int qt   = blockIdx.x * 8 + wave;   // 0..127
  const int bh   = blockIdx.y;              // 0..63
  const int b    = bh >> 4, h = bh & 15;

  const __bf16* Qb = qh + (size_t)bh * SEQ * DEPTH;
  const __bf16* Kb = kh + (size_t)bh * SEQ * DEPTH;
  const __bf16* Vb = vT + (size_t)bh * DEPTH * SEQ;
  const int*    mb = mask + b * SEQ;

  const int m0 = qt * 16;
  const int c0 = (lane >> 4) * 8;
  __bf16* psm = &pshare[wave][0];

  // LDS byte offsets (low 32 bits of a generic pointer to LDS == DS offset).
  const unsigned ks_off  = (unsigned)(size_t)&ksm[0] + (unsigned)t * 16u;
  const unsigned vs_base = (unsigned)(size_t)&vsm[0];

  // Q A-fragments for depth chunks [0,32) and [32,64).
  const __bf16* qp = Qb + (size_t)(m0 + (lane & 15)) * DEPTH;
  v16bf qa0 = cat16(ld8(qp + c0),      ld8(qp + c0 + 16));
  v16bf qa1 = cat16(ld8(qp + 32 + c0), ld8(qp + 32 + c0 + 16));

  float mrow[8], lrow[8];
  v8f acc[4] = {};
#pragma unroll
  for (int r = 0; r < 8; ++r) { mrow[r] = -INFINITY; lrow[r] = 0.0f; }

  for (int kb = 0; kb < SEQ; kb += 32) {
    // --- K chunk: contiguous 4KB, per-lane async global->LDS copies ---
    async_ld16(ks_off, (const void*)(Kb + (size_t)kb * DEPTH + t * 8));
    wait_async();

    // --- V chunk: 2D tile via TDM (wave 0 issues for the whole block) ---
    if (wave == 0) {
      unsigned long long ga = (unsigned long long)(size_t)(Vb + kb);
      v4u g0;
      g0.x = 1u;                                   // count=1, user descriptor
      g0.y = vs_base;                              // lds_addr (bytes)
      g0.z = (unsigned)ga;                         // global_addr[31:0]
      g0.w = (unsigned)(ga >> 32) | (2u << 30);    // addr[56:32] | type=2
      v8i g1;
      g1[0] = (1 << 16);            // workgroup_mask=0, data_size=1 (2 bytes)
      g1[1] = (SEQ << 16);          // barrier_addr=0 | tensor_dim0[15:0]
      g1[2] = (DEPTH << 16);        // tensor_dim0[31:16]=0 | tensor_dim1[15:0]
      g1[3] = (32 << 16);           // tensor_dim1[31:16]=0 | tile_dim0=32
      g1[4] = DEPTH;                // tile_dim1=64 | tile_dim2=0
      g1[5] = SEQ;                  // tensor_dim0_stride[31:0] = 2048 elems
      g1[6] = 0;                    // stride0[47:32]=0 | dim1_stride lo=0
      g1[7] = 0;
      v4i gz4 = {0, 0, 0, 0};
      v8i gz8 = {0, 0, 0, 0, 0, 0, 0, 0};
      __builtin_amdgcn_tensor_load_to_lds(g0, g1, gz4, gz4, gz8, 0);
      __builtin_amdgcn_s_wait_tensorcnt(0);
    }
    __syncthreads();   // K (async) + V (TDM) visible to all waves

    // --- scores: S(16x32) = Q(16x64) @ K^T, two 16-key column tiles ---
    v8f sc[2];
#pragma unroll
    for (int j = 0; j < 2; ++j) {
      const __bf16* kp = ksm + (size_t)(j * 16 + (lane & 15)) * DEPTH;
      v16bf kb0 = cat16(ld8(kp + c0),      ld8(kp + c0 + 16));
      v16bf kb1 = cat16(ld8(kp + 32 + c0), ld8(kp + 32 + c0 + 16));
      v8f tt = {};
      tt = __builtin_amdgcn_wmma_f32_16x16x32_bf16(false, qa0, false, kb0,
                                                   (short)0, tt, false, false);
      tt = __builtin_amdgcn_wmma_f32_16x16x32_bf16(false, qa1, false, kb1,
                                                   (short)0, tt, false, false);
      float bias = (mb[kb + j * 16 + (lane & 15)] == 1) ? -1e30f : 0.0f;
#pragma unroll
      for (int r = 0; r < 8; ++r) sc[j][r] = tt[r] * 0.125f + bias;
    }

    // --- online softmax: per-row max/sum via 16-lane shfl_xor reductions ---
    float scale[8];
#pragma unroll
    for (int r = 0; r < 8; ++r) {
      float tv = fmaxf(sc[0][r], sc[1][r]);
      tv = fmaxf(tv, __shfl_xor(tv, 1));
      tv = fmaxf(tv, __shfl_xor(tv, 2));
      tv = fmaxf(tv, __shfl_xor(tv, 4));
      tv = fmaxf(tv, __shfl_xor(tv, 8));
      float nm = fmaxf(mrow[r], tv);
      scale[r] = __expf(mrow[r] - nm);   // exp(-inf - finite) == 0 first pass
      mrow[r]  = nm;
    }
#pragma unroll
    for (int r = 0; r < 8; ++r) {
      float p0 = __expf(sc[0][r] - mrow[r]);
      float p1 = __expf(sc[1][r] - mrow[r]);
      float rs = p0 + p1;
      rs += __shfl_xor(rs, 1);
      rs += __shfl_xor(rs, 2);
      rs += __shfl_xor(rs, 4);
      rs += __shfl_xor(rs, 8);
      lrow[r] = lrow[r] * scale[r] + rs;
      int prow = r + ((lane >> 4) << 3);
      psm[prow * 32 + (lane & 15)]      = (__bf16)p0;
      psm[prow * 32 + 16 + (lane & 15)] = (__bf16)p1;
    }
#pragma unroll
    for (int j = 0; j < 4; ++j)
#pragma unroll
      for (int r = 0; r < 8; ++r) acc[j][r] *= scale[r];

    __syncthreads();   // order P DS stores before DS A-fragment loads

    // --- ctx += P(16x32) @ V(32x64), V fragments from LDS ---
    v16bf pa = cat16(ld8(psm + (lane & 15) * 32 + c0),
                     ld8(psm + (lane & 15) * 32 + c0 + 16));
#pragma unroll
    for (int j = 0; j < 4; ++j) {
      const __bf16* vp = vsm + (size_t)(j * 16 + (lane & 15)) * 32;
      v16bf vb = cat16(ld8(vp + c0), ld8(vp + c0 + 16));
      acc[j] = __builtin_amdgcn_wmma_f32_16x16x32_bf16(
          false, pa, false, vb, (short)0, acc[j], false, false);
    }
    __syncthreads();   // all ksm/vsm reads done before next chunk overwrites
  }

  // --- epilogue: ctx[b, s, h*64+dep] = acc / l  (bf16, merged-head layout) ---
  float linv[8];
#pragma unroll
  for (int r = 0; r < 8; ++r) linv[r] = 1.0f / lrow[r];
#pragma unroll
  for (int j = 0; j < 4; ++j) {
#pragma unroll
    for (int r = 0; r < 8; ++r) {
      int mr  = r + ((lane >> 4) << 3);
      int s   = m0 + mr;
      int col = h * DEPTH + j * 16 + (lane & 15);
      ctx[((size_t)b * SEQ + s) * DM + col] = (__bf16)(acc[j][r] * linv[r]);
    }
  }
}

// ---------------------------------------------------------------------------
// Kernel 3: out = ctx @ fc^T  (all-bf16 fragments, f32 output), same 2-stage
// software pipeline as the projection GEMM.
// ---------------------------------------------------------------------------
__global__ __launch_bounds__(256) void fc_kernel(
    const __bf16* __restrict__ ctx, const __bf16* __restrict__ fcb,
    float* __restrict__ out)
{
  const int lane = threadIdx.x & 31;
  const int wave = threadIdx.x >> 5;
  const int mt   = blockIdx.x * 8 + wave;   // 0..511
  const int nt   = blockIdx.y;              // 0..15

  const int m0 = mt * 16;
  const int n0 = nt * 64;
  const int c0 = (lane >> 4) * 8;

  const __bf16* Ap = ctx + (size_t)(m0 + (lane & 15)) * DM + c0;
  const __bf16* Bp[4];
#pragma unroll
  for (int j = 0; j < 4; ++j)
    Bp[j] = fcb + (size_t)(n0 + j * 16 + (lane & 15)) * DM + c0;

  v8f acc[4] = {};
  v16bf a0 = cat16(ld8(Ap), ld8(Ap + 16));
  v16bf b0[4];
#pragma unroll
  for (int j = 0; j < 4; ++j) b0[j] = cat16(ld8(Bp[j]), ld8(Bp[j] + 16));

  for (int k0 = 0; k0 < DM; k0 += 64) {
    const int k1 = k0 + 32;
    v16bf a1 = cat16(ld8(Ap + k1), ld8(Ap + k1 + 16));
    v16bf b1[4];
#pragma unroll
    for (int j = 0; j < 4; ++j)
      b1[j] = cat16(ld8(Bp[j] + k1), ld8(Bp[j] + k1 + 16));
#pragma unroll
    for (int j = 0; j < 4; ++j)
      acc[j] = __builtin_amdgcn_wmma_f32_16x16x32_bf16(
          false, a0, false, b0[j], (short)0, acc[j], false, false);

    const int k2 = (k0 + 64 < DM) ? k0 + 64 : 0;   // last prefetch is dummy
    a0 = cat16(ld8(Ap + k2), ld8(Ap + k2 + 16));
#pragma unroll
    for (int j = 0; j < 4; ++j)
      b0[j] = cat16(ld8(Bp[j] + k2), ld8(Bp[j] + k2 + 16));
#pragma unroll
    for (int j = 0; j < 4; ++j)
      acc[j] = __builtin_amdgcn_wmma_f32_16x16x32_bf16(
          false, a1, false, b1[j], (short)0, acc[j], false, false);
  }

#pragma unroll
  for (int j = 0; j < 4; ++j) {
#pragma unroll
    for (int r = 0; r < 8; ++r) {
      int mr  = r + ((lane >> 4) << 3);
      int col = n0 + j * 16 + (lane & 15);
      out[(size_t)(m0 + mr) * DM + col] = acc[j][r];
    }
  }
}

// ---------------------------------------------------------------------------
extern "C" void kernel_launch(void* const* d_in, const int* in_sizes, int n_in,
                              void* d_out, int out_size, void* d_ws, size_t ws_size,
                              hipStream_t stream) {
  // setup_inputs order: v, k, q, mask, wq, wk, wv, fc
  const float* v    = (const float*)d_in[0];
  const float* k    = (const float*)d_in[1];
  const float* q    = (const float*)d_in[2];
  const int*   mask = (const int*)d_in[3];
  const float* wq   = (const float*)d_in[4];
  const float* wk   = (const float*)d_in[5];
  const float* wv   = (const float*)d_in[6];
  const float* fc   = (const float*)d_in[7];
  float* out = (float*)d_out;

  char* ws = (char*)d_ws;
  const size_t SEG  = (size_t)16 * 1024 * 1024;   // 8.39M bf16 elements each
  const size_t WSEG = (size_t)2 * 1024 * 1024;    // 1M bf16 weights each
  __bf16* qhb  = (__bf16*)(ws);
  __bf16* khb  = (__bf16*)(ws + SEG);
  __bf16* vTb  = (__bf16*)(ws + 2 * SEG);
  __bf16* ctxb = (__bf16*)(ws + 3 * SEG);
  __bf16* wqb  = (__bf16*)(ws + 4 * SEG);
  __bf16* wkb  = (__bf16*)(ws + 4 * SEG + WSEG);
  __bf16* wvb  = (__bf16*)(ws + 4 * SEG + 2 * WSEG);
  __bf16* fcb  = (__bf16*)(ws + 4 * SEG + 3 * WSEG);

  dim3 blk(256);
  // 1M elems per matrix / (256 thr * 8 elem) = 512 blocks; 4 matrices
  pack_kernel<<<dim3(512, 4), blk, 0, stream>>>(wq, wk, wv, fc,
                                                wqb, wkb, wvb, fcb);
  // 512 M-tiles / 8 waves = 64 blocks; 16 head tiles; z = {q,k,v}
  proj_kernel<<<dim3(64, 16, 3), blk, 0, stream>>>(q, k, v, wqb, wkb, wvb,
                                                   qhb, khb, vTb);
  // 128 q-tiles / 8 waves = 16 blocks; B*H = 64
  attn_kernel<<<dim3(16, 64), blk, 0, stream>>>(qhb, khb, vTb, mask, ctxb);
  // 512 M-tiles / 8 waves = 64 blocks; 16 N-tiles
  fc_kernel<<<dim3(64, 16), blk, 0, stream>>>(ctxb, fcb, out);
}
